// GraphRNN_53541062312278
// MI455X (gfx1250) — matrix operand order repaired and checked by reading, not compiled, not run on hardware
//
#include <hip/hip_runtime.h>
#include <hip/hip_bf16.h>

typedef __attribute__((ext_vector_type(16))) __bf16 v16bf;
typedef __attribute__((ext_vector_type(8)))  float  v8f;
typedef int v4i __attribute__((vector_size(16)));   // 4 x i32, matches builtin

#define EPSV 1e-7f
constexpr int BB = 8;     // batch
constexpr int LL = 1024;  // sequence length
constexpr int DD = 256;   // input feature dim
constexpr int UU = 256;   // units

#define GAS __attribute__((address_space(1)))
#define LAS __attribute__((address_space(3)))

#if defined(__gfx1250__) && __has_builtin(__builtin_amdgcn_global_load_async_to_lds_b128)
#define HAVE_ASYNC_LDS 1
#endif

__device__ __forceinline__ v8f wmma_bf16(v16bf a, v16bf b, v8f c) {
    // D(16x16,f32) = A(16x32,bf16) x B(32x16,bf16) + C
    return __builtin_amdgcn_wmma_f32_16x16x32_bf16(
        /*neg_a=*/false, a, /*neg_b=*/false, b,
        /*c_mod=*/(short)0, c, /*reuse_a=*/false, /*reuse_b=*/false);
}

#ifdef HAVE_ASYNC_LDS
__device__ __forceinline__ void async_g2l_b128(void* lds, const void* g) {
    // ASYNCcnt-tracked DMA: LDS[dst] = MEM[src], 16 bytes per lane.
    // Param 0 is AS(1) v4i*, param 1 is AS(3) v4i* (per hipcc diagnostic).
    // Generic->AS casts done via integer round-trip: AS3 pointer value is the
    // low 32 bits of the generic LDS address (flat aperture rule).
    __builtin_amdgcn_global_load_async_to_lds_b128(
        (GAS v4i*)(unsigned long long)g,
        (LAS v4i*)(unsigned long long)lds,
        /*offset=*/0, /*cpol=*/0);
}
__device__ __forceinline__ void wait_async0() {
#if __has_builtin(__builtin_amdgcn_s_wait_asynccnt)
    __builtin_amdgcn_s_wait_asynccnt(0);
#else
    asm volatile("s_wait_asynccnt 0" ::: "memory");
#endif
}
#endif

// ---------------------------------------------------------------------------
// Kernel 1: rnorm[b*L+l] = 1 / max(sum_m graph[b,l,m], eps)
// ---------------------------------------------------------------------------
__global__ void norm_kernel(const float* __restrict__ graph,
                            float* __restrict__ rnorm) {
    const int row  = blockIdx.x * (blockDim.x >> 5) + (threadIdx.x >> 5);
    const int lane = threadIdx.x & 31;
    const float* g = graph + (size_t)row * LL;
    float s = 0.f;
    for (int i = lane; i < LL; i += 32) s += g[i];
#pragma unroll
    for (int off = 16; off > 0; off >>= 1) s += __shfl_xor(s, off, 32);
    if (lane == 0) rnorm[row] = 1.0f / fmaxf(s, EPSV);
}

// ---------------------------------------------------------------------------
// Kernel 2: beta16[b,l,u] = bf16( (seq @ E)[b,l,u] * rnorm[b,l] )
// One wave computes a 16x64 strip: 1 A-frag reused over 4 B-frags / 4 accs.
// ---------------------------------------------------------------------------
__global__ void beta_kernel(const float* __restrict__ seq,
                            const float* __restrict__ E,
                            const float* __restrict__ rnorm,
                            __bf16* __restrict__ beta16) {
    __shared__ __bf16 aT[8][16][40];   // [wave][m][k]
    __shared__ __bf16 bT[8][64][40];   // [wave][n][k] (B staged transposed)
    const int w    = threadIdx.x >> 5;
    const int lane = threadIdx.x & 31;
    const int tile = blockIdx.x * 8 + w;       // 2048 strips total
    const int b    = tile >> 8;                // / (64*4)
    const int mt   = (tile >> 2) & 63;
    const int nt   = tile & 3;
    const int l0   = mt * 16;
    const int u0   = nt * 64;
    const int n    = lane & 15;
    const int kh   = lane >> 4;

    v8f acc[4] = {};
    for (int k0 = 0; k0 < DD; k0 += 32) {
        if (k0 + 32 < DD && lane < 16)   // prefetch next A tile rows
            __builtin_prefetch(&seq[(size_t)(b * LL + l0 + lane) * DD + k0 + 32], 0, 0);
        // stage A: 16x32, row-scaled by rnorm
        for (int e = lane; e < 512; e += 32) {
            const int mm = e >> 5, kk = e & 31;
            const float v = seq[(size_t)(b * LL + l0 + mm) * DD + k0 + kk] *
                            rnorm[b * LL + l0 + mm];
            aT[w][mm][kk] = (__bf16)v;
        }
        // stage B transposed: bT[n][k] = E[k0+k][u0+n], 64 columns
        for (int e = lane; e < 2048; e += 32) {
            const int kk = e >> 6, nn = e & 63;
            bT[w][nn][kk] = (__bf16)E[(size_t)(k0 + kk) * UU + u0 + nn];
        }
        v16bf af;
#pragma unroll
        for (int i = 0; i < 8; ++i) {
            af[i]     = aT[w][n][kh * 16 + i];       // K = kh*16 + i
            af[8 + i] = aT[w][n][kh * 16 + 8 + i];   // K = kh*16 + 8 + i
        }
#pragma unroll
        for (int s = 0; s < 4; ++s) {
            v16bf bfr;
#pragma unroll
            for (int i = 0; i < 16; ++i)
                bfr[i] = bT[w][s * 16 + n][kh * 16 + i];
            acc[s] = wmma_bf16(af, bfr, acc[s]);
        }
    }
    const int mb = kh * 8;
#pragma unroll
    for (int s = 0; s < 4; ++s)
#pragma unroll
        for (int r = 0; r < 8; ++r)
            beta16[(size_t)(b * LL + l0 + mb + r) * UU + u0 + s * 16 + n] =
                (__bf16)acc[s][r];
}

// ---------------------------------------------------------------------------
// Kernel 3: agg16[b,m,u] = bf16( sum_l graph[b,l,m] * beta[b,l,u] )
// Same 16x64 register blocking; A = graph^T staged transposed into LDS.
// ---------------------------------------------------------------------------
__global__ void agg_kernel(const float* __restrict__ graph,
                           const __bf16* __restrict__ beta16,
                           __bf16* __restrict__ agg16) {
    __shared__ __bf16 aT[8][16][40];   // [wave][m][k], A[m][k]=graph[b][k][m]
    __shared__ __bf16 bT[8][64][40];   // [wave][n][k]
    const int w    = threadIdx.x >> 5;
    const int lane = threadIdx.x & 31;
    const int tile = blockIdx.x * 8 + w;
    const int b    = tile >> 8;
    const int mt   = (tile >> 2) & 63;
    const int nt   = tile & 3;
    const int m0   = mt * 16;
    const int u0   = nt * 64;
    const int n    = lane & 15;
    const int kh   = lane >> 4;

    v8f acc[4] = {};
    for (int k0 = 0; k0 < LL; k0 += 32) {
        if (k0 + 32 < LL)   // prefetch next 32 graph rows (one lane each)
            __builtin_prefetch(
                &graph[((size_t)b * LL + k0 + 32 + lane) * LL + m0], 0, 0);
        // stage transposed A: aT[m][k] = graph[b][k0+k][m0+m]
        for (int e = lane; e < 512; e += 32) {
            const int kk = e >> 4, mm = e & 15;
            aT[w][mm][kk] =
                (__bf16)graph[((size_t)b * LL + k0 + kk) * LL + m0 + mm];
        }
        // stage B transposed: bT[n][k] = beta16[b][k0+k][u0+n]
        for (int e = lane; e < 2048; e += 32) {
            const int kk = e >> 6, nn = e & 63;
            bT[w][nn][kk] = beta16[(size_t)(b * LL + k0 + kk) * UU + u0 + nn];
        }
        v16bf af;
#pragma unroll
        for (int i = 0; i < 8; ++i) {
            af[i]     = aT[w][n][kh * 16 + i];
            af[8 + i] = aT[w][n][kh * 16 + 8 + i];
        }
#pragma unroll
        for (int s = 0; s < 4; ++s) {
            v16bf bfr;
#pragma unroll
            for (int i = 0; i < 16; ++i)
                bfr[i] = bT[w][s * 16 + n][kh * 16 + i];
            acc[s] = wmma_bf16(af, bfr, acc[s]);
        }
    }
    const int mb = kh * 8;
#pragma unroll
    for (int s = 0; s < 4; ++s)
#pragma unroll
        for (int r = 0; r < 8; ++r)
            agg16[(size_t)(b * LL + m0 + mb + r) * UU + u0 + s * 16 + n] =
                (__bf16)acc[s][r];
}

// ---------------------------------------------------------------------------
// Kernel 4: sequential RNN scan, persistent single workgroup (16 waves).
// Wx^T/Wh^T resident in LDS (bf16, ~280KB dynamic LDS — CDNA5 320KB/WGP).
// Weight B-fragments hoisted into registers (invariant over 1024 steps).
// x_l staged via async global->LDS DMA when the toolchain exposes it.
// h' = tanh(x @ Wx + h @ Wh + b)
// ---------------------------------------------------------------------------
__global__ void __launch_bounds__(512)
rnn_kernel(const __bf16* __restrict__ agg16,
           const float* __restrict__ Wx,
           const float* __restrict__ Wh,
           const float* __restrict__ bias,
           float* __restrict__ out) {
    extern __shared__ __align__(16) char smem[];
    __bf16* WxT = (__bf16*)smem;           // [U][U]: WxT[u][k] = Wx[k][u]
    __bf16* WhT = WxT + UU * UU;           // [U][U]
    __bf16* Xb  = WhT + UU * UU;           // [16][U]
    __bf16* Hb  = Xb + 16 * UU;            // 2 x [16][U] double buffer

    const int tid  = threadIdx.x;
    const int w    = tid >> 5;
    const int lane = tid & 31;
    const int u0   = w * 16;
    const int n    = lane & 15;
    const int kh   = lane >> 4;
    const int m    = lane & 15;            // A-fragment row

    // prologue: transpose+convert weights into LDS, zero x/h buffers
    for (int e = tid; e < UU * UU; e += 512) {
        const int k = e >> 8, u = e & 255;
        WxT[u * UU + k] = (__bf16)Wx[e];
        WhT[u * UU + k] = (__bf16)Wh[e];
    }
    for (int e = tid; e < 16 * UU; e += 512) Xb[e] = (__bf16)0.f;
    for (int e = tid; e < 32 * UU; e += 512) Hb[e] = (__bf16)0.f;
    const float bv = bias[u0 + n];
    __syncthreads();

    // hoist weight B-fragments (loop-invariant across all 1024 steps)
    v16bf bx[8], bh[8];
#pragma unroll
    for (int t = 0; t < 8; ++t) {
        const int k0 = t * 32;
#pragma unroll
        for (int i = 0; i < 16; ++i) {
            bx[t][i] = WxT[(u0 + n) * UU + k0 + kh * 16 + i];
            bh[t][i] = WhT[(u0 + n) * UU + k0 + kh * 16 + i];
        }
    }

    for (int l = 0; l < LL; ++l) {
        // stage x_l (rows 0..7 = real batches; rows 8..15 stay zero)
#ifdef HAVE_ASYNC_LDS
        if (tid < 256) {                        // 256 x 16B = 4KB DMA
            const int bb = tid >> 5;            // batch row
            const int c  = tid & 31;            // 16B chunk within row
            async_g2l_b128(&Xb[bb * UU + c * 8],
                           agg16 + (size_t)(bb * LL + l) * UU + c * 8);
        }
        wait_async0();
#else
        for (int e = tid; e < 8 * UU; e += 512) {
            const int bb = e >> 8, u = e & 255;
            Xb[bb * UU + u] = agg16[(size_t)(bb * LL + l) * UU + u];
        }
#endif
        __syncthreads();

        const __bf16* Hc = Hb + (l & 1) * 16 * UU;
        __bf16*       Hn = Hb + ((l + 1) & 1) * 16 * UU;

        v8f acc;
#pragma unroll
        for (int i = 0; i < 8; ++i) acc[i] = bv;   // + bias

#pragma unroll
        for (int t = 0; t < 8; ++t) {
            const int k0 = t * 32;
            v16bf ax, ah;
#pragma unroll
            for (int i = 0; i < 8; ++i) {
                ax[i]     = Xb[m * UU + k0 + kh * 16 + i];
                ax[8 + i] = Xb[m * UU + k0 + kh * 16 + 8 + i];
                ah[i]     = Hc[m * UU + k0 + kh * 16 + i];
                ah[8 + i] = Hc[m * UU + k0 + kh * 16 + 8 + i];
            }
            acc = wmma_bf16(ax, bx[t], acc);
            acc = wmma_bf16(ah, bh[t], acc);
        }

        // tanh, write next-h (disjoint buffer from Hc), emit fp32 output
#pragma unroll
        for (int r = 0; r < 8; ++r) {
            const float v  = tanhf(acc[r]);
            const int mrow = kh * 8 + r;
            Hn[mrow * UU + u0 + n] = (__bf16)v;
            if (mrow < BB)
                out[((size_t)mrow * LL + l) * UU + u0 + n] = v;
        }
        __syncthreads();   // Hn visible & Xb reads done before next stage
    }
}

// ---------------------------------------------------------------------------
extern "C" void kernel_launch(void* const* d_in, const int* in_sizes, int n_in,
                              void* d_out, int out_size, void* d_ws,
                              size_t ws_size, hipStream_t stream) {
    (void)in_sizes; (void)n_in; (void)out_size; (void)ws_size;
    const float* seq   = (const float*)d_in[0];  // (B,L,D)
    const float* graph = (const float*)d_in[1];  // (B,L,L)
    const float* E     = (const float*)d_in[2];  // (D,U)
    const float* Wx    = (const float*)d_in[3];  // (U,U)
    const float* Wh    = (const float*)d_in[4];  // (U,U)
    const float* bias  = (const float*)d_in[5];  // (U,)
    float* out = (float*)d_out;                  // (B,L,U) fp32

    char* ws = (char*)d_ws;
    float*  rnorm  = (float*)ws;                                   // 32 KB
    __bf16* beta16 = (__bf16*)(ws + 64 * 1024);                    // 4 MB
    __bf16* agg16  = (__bf16*)(ws + 64 * 1024 +
                               (size_t)BB * LL * UU * sizeof(__bf16));

    norm_kernel<<<(BB * LL) / 8, 256, 0, stream>>>(graph, rnorm);

    const int strips = BB * (LL / 16) * (UU / 64);   // 2048, 8 waves/block
    beta_kernel<<<strips / 8, 256, 0, stream>>>(seq, E, rnorm, beta16);
    agg_kernel<<<strips / 8, 256, 0, stream>>>(graph, beta16, agg16);

    const size_t shm =
        (size_t)(2 * UU * UU + 16 * UU + 32 * UU) * sizeof(__bf16); // ~280 KB
    (void)hipFuncSetAttribute(reinterpret_cast<const void*>(rnn_kernel),
                              hipFuncAttributeMaxDynamicSharedMemorySize,
                              (int)shm);
    rnn_kernel<<<1, 512, shm, stream>>>(agg16, Wx, Wh, bias, out);
}
